// MultiheadAttentionWithRoPE_74646531605195
// MI455X (gfx1250) — matrix operand order repaired and checked by reading, not compiled, not run on hardware
//
#include <hip/hip_runtime.h>
#include <hip/hip_bf16.h>

// ---------------------------------------------------------------------------
// MI455X (gfx1250) fused MHA + RoPE.
// - all matmuls via v_wmma_f32_16x16x32_f16 (f16 in, f32 accumulate)
// - double-buffered tile staging via global_load_async_to_lds_b128, with
//   s_wait_asynccnt <= N so the in-flight copy overlaps the WMMA compute
// ---------------------------------------------------------------------------

typedef __attribute__((ext_vector_type(16))) _Float16 v16h;
typedef __attribute__((ext_vector_type(8)))  float    v8f;

union Frag16 {              // one WMMA f16 operand (8 VGPRs / 16 halves per lane)
    v16h      v;
    uint4     q[2];
    _Float16  h[16];
};
union Acc8 {                // one WMMA f32 C/D tile (8 VGPRs per lane)
    v8f   v;
    float f[8];
};
union H2 {
    _Float16 h[2];
    unsigned int u;
};

#define WMMA16(a, b, c) \
    __builtin_amdgcn_wmma_f32_16x16x32_f16(false, (a), false, (b), (short)0, (c), false, false)

// async global -> LDS copy of 16 bytes per lane (CDNA5 ASYNCcnt path)
__device__ __forceinline__ void async_copy_b128(void* lds_dst, const void* gsrc) {
    unsigned lds_off = (unsigned)(unsigned long long)(uintptr_t)lds_dst;
    asm volatile("global_load_async_to_lds_b128 %0, %1, off"
                 :: "v"(lds_off), "v"(gsrc) : "memory");
}
// wait until at most N async ops remain outstanding (in-order completion)
#define WAIT_ASYNC_LE(n) asm volatile("s_wait_asynccnt " #n ::: "memory")

static constexpr int Bn = 4;
static constexpr int Tn = 2048;
static constexpr int Hn = 16;
static constexpr int Dn = 64;
static constexpr int DM = 1024;
static constexpr int N3 = 3072;
static constexpr int Mrows = Bn * Tn;   // 8192

// ---------------------------------------------------------------------------
// elementwise f32 -> f16 cast
// ---------------------------------------------------------------------------
__global__ void cast_f32_f16_kernel(const float* __restrict__ in,
                                    _Float16* __restrict__ out, long long n) {
    long long stride = (long long)gridDim.x * blockDim.x;
    for (long long i = (long long)blockIdx.x * blockDim.x + threadIdx.x; i < n; i += stride)
        out[i] = (_Float16)in[i];
}

// in: [K][N] f32 row-major  ->  out: [N][K] f16 (transposed, K contiguous)
__global__ void transpose_cast_kernel(const float* __restrict__ in,
                                      _Float16* __restrict__ out, int K, int N) {
    long long total = (long long)K * N;
    long long stride = (long long)gridDim.x * blockDim.x;
    for (long long i = (long long)blockIdx.x * blockDim.x + threadIdx.x; i < total; i += stride) {
        long long n = i / K;
        long long k = i - n * K;
        out[i] = (_Float16)in[k * (long long)N + n];
    }
}

// ---------------------------------------------------------------------------
// LDS-tiled WMMA GEMM:  C[M][Ntot] = A[M][1024] * Bt[Ntot][1024]^T + bias
// Workgroup: 256 threads = 8 waves -> 64(M) x 128(N) tile, K-step 32.
// Double-buffered async global->LDS staging (3 async b128 per thread/step).
// ---------------------------------------------------------------------------
template <int NTOT, bool F16OUT>
__global__ void wmma_gemm_kernel(const _Float16* __restrict__ A,
                                 const _Float16* __restrict__ Bt,
                                 const float* __restrict__ bias,
                                 void* __restrict__ Cout) {
    constexpr int K = 1024;
    __shared__ __align__(16) _Float16 As[2][64 * 32];
    __shared__ __align__(16) _Float16 Bs[2][128 * 32];

    const int tid  = threadIdx.x;
    const int lane = tid & 31;
    const int wave = tid >> 5;
    const int mw   = wave & 3;
    const int nw   = wave >> 2;
    const int hb   = lane >> 4;      // half-wave group (0/1)
    const int l15  = lane & 15;
    const int mblock = blockIdx.x * 64;
    const int nblock = blockIdx.y * 128;

    // staging assignments
    const int arow = tid >> 2, achunk = tid & 3;   // 64 rows x 4x(8 halves)
    const int brow = tid >> 1, bhalf  = tid & 1;   // 128 rows x 2x(16 halves)
    const _Float16* ag = &A[(size_t)(mblock + arow) * K + achunk * 8];
    const _Float16* bg = &Bt[(size_t)(nblock + brow) * K + bhalf * 16];
    const int a_off  = arow * 32 + achunk * 8;
    const int b_off0 = brow * 32 + bhalf * 16;
    const int b_off1 = b_off0 + 8;

    Acc8 acc[4];
#pragma unroll
    for (int i = 0; i < 4; ++i)
#pragma unroll
        for (int r = 0; r < 8; ++r) acc[i].f[r] = 0.0f;

    // prologue: fill buffer 0 with the k0 = 0 tiles
    async_copy_b128(&As[0][a_off],  ag);
    async_copy_b128(&Bs[0][b_off0], bg);
    async_copy_b128(&Bs[0][b_off1], bg + 8);

    int buf = 0;
    for (int k0 = 0; k0 < K; k0 += 32) {
        const int nxt = buf ^ 1;
        const int kn  = (k0 + 32) & (K - 1);     // wrapped prefetch (last iter: dummy)
        __syncthreads();                          // nobody still reads buffer 'nxt'
        async_copy_b128(&As[nxt][a_off],  ag + kn);
        async_copy_b128(&Bs[nxt][b_off0], bg + kn);
        async_copy_b128(&Bs[nxt][b_off1], bg + kn + 8);
        WAIT_ASYNC_LE(0x3);                       // drain this buffer's 3 copies
        __syncthreads();                          // all threads' fills visible

        // A fragment (16x32): lane row = l15; halves K = hb*8+[0,8) and 16+hb*8+[0,8)
        Frag16 af;
        const int am = mw * 16 + l15;
        af.q[0] = *(const uint4*)&As[buf][am * 32 + hb * 8];
        af.q[1] = *(const uint4*)&As[buf][am * 32 + 16 + hb * 8];
#pragma unroll
        for (int i = 0; i < 4; ++i) {
            // B fragment (32x16): lane col = l15; halves K = hb*8+[0,8) and 16+hb*8+[0,8)
            Frag16 bf;
            const int bn = nw * 64 + i * 16 + l15;
            bf.q[0] = *(const uint4*)&Bs[buf][bn * 32 + hb * 8];
            bf.q[1] = *(const uint4*)&Bs[buf][bn * 32 + 16 + hb * 8];
            acc[i].v = WMMA16(af.v, bf.v, acc[i].v);
        }
        buf = nxt;
    }

    // epilogue: C layout -> VGPR r: m_local = r + hb*8, n_local = l15
#pragma unroll
    for (int i = 0; i < 4; ++i) {
        const int n = nblock + nw * 64 + i * 16 + l15;
        const float bv = bias[n];
#pragma unroll
        for (int r = 0; r < 8; ++r) {
            const int m = mblock + mw * 16 + r + hb * 8;
            const float val = acc[i].f[r] + bv;
            if (F16OUT)
                ((_Float16*)Cout)[(size_t)m * NTOT + n] = (_Float16)val;
            else
                ((float*)Cout)[(size_t)m * NTOT + n] = val;
        }
    }
}

// ---------------------------------------------------------------------------
// RoPE on q,k + repack: qkvh[b*T+t][3*1024] -> Qh,Kh [B*H][T][D], Vt [B*H][D][T]
// The 1/sqrt(64) attention scale is folded into Q here.
// ---------------------------------------------------------------------------
__global__ void rope_layout_kernel(const _Float16* __restrict__ qkvh,
                                   _Float16* __restrict__ Qh,
                                   _Float16* __restrict__ Kh,
                                   _Float16* __restrict__ Vt) {
    const int idx = blockIdx.x * blockDim.x + threadIdx.x;   // bh*T + t
    if (idx >= Bn * Hn * Tn) return;
    const int t  = idx & (Tn - 1);
    const int bh = idx >> 11;
    const int b  = bh >> 4;
    const int h  = bh & 15;

    const _Float16* src  = qkvh + ((size_t)(b * Tn + t) * N3) + h * Dn;
    const _Float16* ksrc = src + DM;
    const _Float16* vsrc = src + 2 * DM;
    _Float16* qo = Qh + (size_t)idx * Dn;
    _Float16* ko = Kh + (size_t)idx * Dn;

    const float qscale = 0.125f;   // 1/sqrt(HEAD_DIM)

#pragma unroll 4
    for (int j = 0; j < 32; ++j) {
        const float ang = (float)t * __powf(10000.0f, (float)(-2 * j) * (1.0f / 64.0f));
        float sn, cs;
        __sincosf(ang, &sn, &cs);
        const float q1 = (float)src[2 * j], q2 = (float)src[2 * j + 1];
        qo[j]      = (_Float16)((q1 * cs - q2 * sn) * qscale);
        qo[32 + j] = (_Float16)((q1 * sn + q2 * cs) * qscale);
        const float k1 = (float)ksrc[2 * j], k2 = (float)ksrc[2 * j + 1];
        ko[j]      = (_Float16)(k1 * cs - k2 * sn);
        ko[32 + j] = (_Float16)(k1 * sn + k2 * cs);
    }
    // V transposed: Vt[(bh*64 + d)*T + t]  (lanes = adjacent t -> coalesced)
    _Float16* vo = Vt + (size_t)bh * Dn * Tn + t;
#pragma unroll 8
    for (int d = 0; d < Dn; ++d) vo[(size_t)d * Tn] = vsrc[d];
}

// ---------------------------------------------------------------------------
// Flash attention. One wave owns 16 queries; 8 waves of the WG share (b,h)
// and stream the same key blocks, so K/V tiles are staged once per WG into
// double-buffered LDS with async copies (copy of block kb+32 overlaps the
// WMMA compute of block kb). Computes S^T = K*Q^T and O^T = V^T*P^T so
// online softmax and the P^T repack need only xor-16 shuffles.
// grid = (B*H, T/128), block = 256.
// ---------------------------------------------------------------------------
__global__ void flash_attn_kernel(const _Float16* __restrict__ Qh,
                                  const _Float16* __restrict__ Kh,
                                  const _Float16* __restrict__ Vt,
                                  const unsigned char* __restrict__ mask,
                                  _Float16* __restrict__ Oh) {
    __shared__ __align__(16) _Float16 Ks[2][32 * 64];   // 32 keys x 64 d
    __shared__ __align__(16) _Float16 Vs[2][64 * 32];   // 64 d x 32 keys (V^T)

    const int tid  = threadIdx.x;
    const int bh   = blockIdx.x;
    const int b    = bh >> 4;
    const int h    = bh & 15;
    const int wave = tid >> 5;
    const int lane = tid & 31;
    const int hb   = lane >> 4;
    const int l15  = lane & 15;
    const int qb   = blockIdx.y * 128 + wave * 16;

    const _Float16* Qbase = Qh + (size_t)bh * Tn * Dn;
    const _Float16* Kbase = Kh + (size_t)bh * Tn * Dn;
    const _Float16* Vbase = Vt + (size_t)bh * Dn * Tn;
    const unsigned char* mrow = mask + (size_t)b * Tn;

    // staging assignments (256 threads, 16B each)
    const int krow_s = tid >> 3, kchunk = tid & 7;   // 32 rows x 8x(8 halves)
    const int vrow_s = tid >> 2, vchunk = tid & 3;   // 64 rows x 4x(8 halves)
    const _Float16* kg = Kbase + (size_t)krow_s * Dn + kchunk * 8;
    const _Float16* vg = Vbase + (size_t)vrow_s * Tn + vchunk * 8;
    const int k_off = krow_s * 64 + kchunk * 8;
    const int v_off = vrow_s * 32 + vchunk * 8;

    // Q^T B-operand, 2 d-chunks: lane col = query l15, halves K = d = 32c+hb*8+[0,8) and 32c+16+hb*8+[0,8)
    Frag16 qf[2];
    {
        const _Float16* qrow = Qbase + (size_t)(qb + l15) * Dn;
#pragma unroll
        for (int c = 0; c < 2; ++c) {
            qf[c].q[0] = *(const uint4*)&qrow[c * 32 + hb * 8];
            qf[c].q[1] = *(const uint4*)&qrow[c * 32 + 16 + hb * 8];
        }
    }

    Acc8 o[4];
#pragma unroll
    for (int i = 0; i < 4; ++i)
#pragma unroll
        for (int r = 0; r < 8; ++r) o[i].f[r] = 0.0f;
    float mrun = -1e30f, drun = 0.0f;

    // prologue: fill buffer 0 with key block 0
    async_copy_b128(&Ks[0][k_off], kg);
    async_copy_b128(&Vs[0][v_off], vg);

    int buf = 0;
    for (int kb = 0; kb < Tn; kb += 32) {
        const int nxt = buf ^ 1;
        const int kn  = (kb + 32) & (Tn - 1);    // wrapped prefetch (last iter: dummy)
        __syncthreads();                          // nobody still reads buffer 'nxt'
        async_copy_b128(&Ks[nxt][k_off], kg + (size_t)kn * Dn);
        async_copy_b128(&Vs[nxt][v_off], vg + kn);
        WAIT_ASYNC_LE(0x2);                       // drain this buffer's 2 copies
        __syncthreads();                          // all threads' fills visible

        // --- S^T = K * Q^T : A = K-tile (16 keys x 32 d), 2 key tiles x 2 d-chunks
        Acc8 st[2];
#pragma unroll
        for (int tt = 0; tt < 2; ++tt) {
#pragma unroll
            for (int r = 0; r < 8; ++r) st[tt].f[r] = 0.0f;
#pragma unroll
            for (int c = 0; c < 2; ++c) {
                Frag16 kf;
                const int krow = tt * 16 + l15;
                kf.q[0] = *(const uint4*)&Ks[buf][krow * 64 + c * 32 + hb * 8];
                kf.q[1] = *(const uint4*)&Ks[buf][krow * 64 + c * 32 + 16 + hb * 8];
                st[tt].v = WMMA16(kf.v, qf[c].v, st[tt].v);
            }
        }

        // padding mask (rare): lane holds keys kj = tt*16 + r + hb*8
        {
            const unsigned long long mm0 =
                *(const unsigned long long*)&mrow[kb + hb * 8];
            const unsigned long long mm1 =
                *(const unsigned long long*)&mrow[kb + 16 + hb * 8];
            if (mm0 | mm1) {
#pragma unroll
                for (int r = 0; r < 8; ++r) {
                    if ((mm0 >> (8 * r)) & 0xffull) st[0].f[r] = -1e30f;
                    if ((mm1 >> (8 * r)) & 0xffull) st[1].f[r] = -1e30f;
                }
            }
        }

        // --- online softmax (per lane = per query; xor-16 pairs the key halves)
        float bm = -1e30f;
#pragma unroll
        for (int r = 0; r < 8; ++r) bm = fmaxf(bm, fmaxf(st[0].f[r], st[1].f[r]));
        bm = fmaxf(bm, __shfl_xor(bm, 16, 32));
        const float nm    = fmaxf(mrun, bm);
        const float alpha = __expf(mrun - nm);
        float p0[8], p1[8], lsum = 0.0f;
#pragma unroll
        for (int r = 0; r < 8; ++r) {
            p0[r] = __expf(st[0].f[r] - nm);
            p1[r] = __expf(st[1].f[r] - nm);
            lsum += p0[r] + p1[r];
        }
        lsum += __shfl_xor(lsum, 16, 32);
        drun = drun * alpha + lsum;
        mrun = nm;
#pragma unroll
        for (int i = 0; i < 4; ++i)
#pragma unroll
            for (int r = 0; r < 8; ++r) o[i].f[r] *= alpha;

        // --- repack P^T (C layout) into B-operand via xor-16 shuffles
        float o0[8], o1[8];
#pragma unroll
        for (int r = 0; r < 8; ++r) {
            o0[r] = __shfl_xor(p0[r], 16, 32);
            o1[r] = __shfl_xor(p1[r], 16, 32);
        }
        Frag16 pv;
        const bool hi = (hb != 0);
#pragma unroll
        for (int j = 0; j < 8; ++j) {
            pv.h[j]     = (_Float16)(hi ? o1[j] : p0[j]);   // K = j      (+16 if hi)
            pv.h[8 + j] = (_Float16)(hi ? p1[j] : o0[j]);   // K = 8 + j  (+16 if hi)
        }

        // --- O^T += V^T * P^T : A = V^T tile (16 d x 32 keys) from LDS
#pragma unroll
        for (int tt = 0; tt < 4; ++tt) {
            Frag16 vf;
            const int vrow = tt * 16 + l15;
            vf.q[0] = *(const uint4*)&Vs[buf][vrow * 32 + hb * 16];
            vf.q[1] = *(const uint4*)&Vs[buf][vrow * 32 + hb * 16 + 8];
            o[tt].v = WMMA16(vf.v, pv.v, o[tt].v);
        }
        buf = nxt;
    }

    // normalize and store O (row = query qb+l15, cols d = tt*16 + hb*8 + r)
    const float inv = 1.0f / drun;
    _Float16* orow = Oh + (size_t)(b * Tn + qb + l15) * DM + h * Dn;
#pragma unroll
    for (int tt = 0; tt < 4; ++tt) {
#pragma unroll
        for (int r = 0; r < 8; r += 2) {
            H2 p;
            p.h[0] = (_Float16)(o[tt].f[r] * inv);
            p.h[1] = (_Float16)(o[tt].f[r + 1] * inv);
            *(unsigned int*)&orow[tt * 16 + hb * 8 + r] = p.u;
        }
    }
}

// ---------------------------------------------------------------------------
extern "C" void kernel_launch(void* const* d_in, const int* in_sizes, int n_in,
                              void* d_out, int out_size, void* d_ws, size_t ws_size,
                              hipStream_t stream) {
    const float* x             = (const float*)d_in[0];
    const unsigned char* pmask = (const unsigned char*)d_in[1];
    const float* Wqkv          = (const float*)d_in[2];
    const float* bqkv          = (const float*)d_in[3];
    const float* Wo            = (const float*)d_in[4];
    const float* bo            = (const float*)d_in[5];
    float* out                 = (float*)d_out;

    char* ws = (char*)d_ws;
    size_t off = 0;
    _Float16* xh    = (_Float16*)(ws + off); off += (size_t)Mrows * DM * 2;
    _Float16* wqkvt = (_Float16*)(ws + off); off += (size_t)N3 * DM * 2;
    _Float16* wot   = (_Float16*)(ws + off); off += (size_t)DM * DM * 2;
    const size_t qkv_off = off;
    _Float16* qkvh  = (_Float16*)(ws + off); off += (size_t)Mrows * N3 * 2;
    _Float16* Qh    = (_Float16*)(ws + off); off += (size_t)Bn * Hn * Tn * Dn * 2;
    _Float16* Kh    = (_Float16*)(ws + off); off += (size_t)Bn * Hn * Tn * Dn * 2;
    _Float16* Vt    = (_Float16*)(ws + off); off += (size_t)Bn * Hn * Tn * Dn * 2;
    _Float16* Oh    = (_Float16*)(ws + qkv_off);   // alias: qkvh dead after rope

    cast_f32_f16_kernel<<<4096, 256, 0, stream>>>(x, xh, (long long)Mrows * DM);
    transpose_cast_kernel<<<4096, 256, 0, stream>>>(Wqkv, wqkvt, DM, N3);
    transpose_cast_kernel<<<2048, 256, 0, stream>>>(Wo, wot, DM, DM);

    wmma_gemm_kernel<N3, true>
        <<<dim3(Mrows / 64, N3 / 128), 256, 0, stream>>>(xh, wqkvt, bqkv, (void*)qkvh);

    rope_layout_kernel<<<(Bn * Hn * Tn) / 256, 256, 0, stream>>>(qkvh, Qh, Kh, Vt);

    flash_attn_kernel<<<dim3(Bn * Hn, Tn / 128), 256, 0, stream>>>(Qh, Kh, Vt, pmask, Oh);

    wmma_gemm_kernel<DM, false>
        <<<dim3(Mrows / 64, DM / 128), 256, 0, stream>>>(Oh, wot, bo, (void*)out);
}
